// Net_35416300322976
// MI455X (gfx1250) — matrix operand order repaired and checked by reading, not compiled, not run on hardware
//
#include <hip/hip_runtime.h>
#include <hip/hip_bf16.h>

// ---------------------------------------------------------------------------
// SplineCNN (superpixel GNN) forward for MI455X / gfx1250.
// Dense contractions: v_wmma_f32_16x16x32_f16 (wave32 WMMA, f32 accumulate).
// Edge scatter buckets: f16 T tensor via global_atomic_pk_add_f16 (packed
// channel pairs -> half the atomic traffic of f32), f32 path for cin==1.
// Segment-max pooling: ordered-uint atomicMax.
// ---------------------------------------------------------------------------

typedef _Float16 v16h __attribute__((ext_vector_type(16)));
typedef _Float16 h8v  __attribute__((ext_vector_type(8)));
typedef _Float16 h4v  __attribute__((ext_vector_type(4)));
typedef float    v8f  __attribute__((ext_vector_type(8)));

#define TILE_M 64
#define TILE_N 64
#define TILE_K 32
#define LDS_STRIDE 48   // halves; 96B row stride keeps 16B-aligned h8 loads

// ---- ordered-float encoding for atomicMax on floats ----
__device__ __forceinline__ unsigned fenc(float f) {
    unsigned u = __float_as_uint(f);
    return (u & 0x80000000u) ? ~u : (u | 0x80000000u);
}
__device__ __forceinline__ float fdec(unsigned u) {
    return (u & 0x80000000u) ? __uint_as_float(u & 0x7fffffffu)
                             : __uint_as_float(~u);
}
__device__ __forceinline__ float elu1(float v) {
    return v > 0.0f ? v : (__expf(v) - 1.0f);
}

// packed f16 atomic add: GLOBAL_ATOMIC_PK_ADD_F16 (CDNA5)
__device__ __forceinline__ void atomic_pk_add_f16(_Float16* addr, float lo, float hi) {
    union { _Float16 h[2]; unsigned u; } pk;
    pk.h[0] = (_Float16)lo;
    pk.h[1] = (_Float16)hi;
    asm volatile("global_atomic_pk_add_f16 %0, %1, off"
                 :: "v"((unsigned long long)(uintptr_t)addr), "v"(pk.u)
                 : "memory");
}

// ---------------------------------------------------------------------------
// Generic WMMA GEMM: C[M,N] = A[M,K] @ B[K,N]  (A f32 or f16 in memory, B f32,
// f16 MACs, f32 accumulate).  Epilogue (all optional):
//   v = acc
//   if (Cprev)  v += Cprev[m,n] * (deg ? 1/max(deg[m],1) : 1)     // mean aggr
//   if (bias)   v += bias[n]
//   if (bng)    v  = v*bng[n] + bnb[n]                            // BN (eval)
//   if (elu)    v  = ELU(v)
// Block = 128 threads = 4 waves; each wave owns a 32x32 quadrant (4 WMMAs/step).
// ---------------------------------------------------------------------------
template <typename AT>
__launch_bounds__(128)
__global__ void wmma_gemm(const AT* __restrict__ A, const float* __restrict__ B,
                          float* __restrict__ C, int M, int N, int K,
                          const float* __restrict__ Cprev, const float* __restrict__ deg,
                          const float* __restrict__ bias, const float* __restrict__ bng,
                          const float* __restrict__ bnb, int elu)
{
    __shared__ __align__(16) _Float16 ldsA[TILE_M * LDS_STRIDE];   // [m][k]
    __shared__ __align__(16) _Float16 ldsB[TILE_N * LDS_STRIDE];   // [n][k] (transposed)

    const int tid   = threadIdx.x;
    const int lane  = tid & 31;
    const int wave  = tid >> 5;
    const int tileM = blockIdx.y * TILE_M;
    const int tileN = blockIdx.x * TILE_N;
    const int wm    = (wave >> 1) * 32;
    const int wn    = (wave & 1) * 32;

    const bool fullM = (tileM + TILE_M <= M);
    const bool fullN = (tileN + TILE_N <= N);
    constexpr int AVEC = (sizeof(AT) == 2) ? 8 : 4;   // elems per 16B A load
    constexpr int NA = (TILE_M * TILE_K) / 128;       // 16 elems/thread (slow path)
    constexpr int NBq = (TILE_K * TILE_N) / 128;      // 16 elems/thread (slow path)

    v8f acc[2][2] = {};

    for (int kb = 0; kb < K; kb += TILE_K) {
        const bool fullK = (kb + TILE_K <= K);

        // ---------------- stage A tile (64 x 32) -> f16 LDS ----------------
        if (fullM && fullK && ((K & (AVEC - 1)) == 0)) {
            if constexpr (sizeof(AT) == 2) {
                // pure 16B copy: 2 x h8 per thread
                const int c8 = (tid & 3) * 8;
                const int r0 = tid >> 2;            // 0..31
#pragma unroll
                for (int p = 0; p < 2; ++p) {
                    int r = r0 + p * 32;
                    h8v v = *(const h8v*)&A[(size_t)(tileM + r) * K + kb + c8];
                    *(h8v*)&ldsA[r * LDS_STRIDE + c8] = v;
                }
            } else {
                // 4 x float4 per thread, convert, b64 LDS stores
                const int c4 = (tid & 7) * 4;
                const int r0 = tid >> 3;            // 0..15
#pragma unroll
                for (int p = 0; p < 4; ++p) {
                    int r = r0 + p * 16;
                    float4 v = *(const float4*)&A[(size_t)(tileM + r) * K + kb + c4];
                    h4v h = { (_Float16)v.x, (_Float16)v.y, (_Float16)v.z, (_Float16)v.w };
                    *(h4v*)&ldsA[r * LDS_STRIDE + c4] = h;
                }
            }
        } else {
            // two-phase: batch all clamped loads into regs, then cvt+store
            float va[NA];
#pragma unroll
            for (int u = 0; u < NA; ++u) {
                int i = tid + u * 128;
                int r = i >> 5, c = i & 31;
                int gm = min(tileM + r, M - 1), gk = min(kb + c, K - 1);
                va[u] = (float)A[(size_t)gm * K + gk];
            }
#pragma unroll
            for (int u = 0; u < NA; ++u) {
                int i = tid + u * 128;
                int r = i >> 5, c = i & 31;
                bool ok = (tileM + r < M) && (kb + c < K);
                ldsA[r * LDS_STRIDE + c] = ok ? (_Float16)va[u] : (_Float16)0.0f;
            }
        }

        // ------------- stage B tile (32 x 64) transposed [n][k] -------------
        if (fullN && fullK && ((N & 3) == 0)) {
            const int c4 = (tid & 15) * 4;          // n quad
            const int k0 = tid >> 4;                // 0..7
#pragma unroll
            for (int p = 0; p < 4; ++p) {
                int k = k0 + p * 8;
                float4 v = *(const float4*)&B[(size_t)(kb + k) * N + tileN + c4];
                ldsB[(c4 + 0) * LDS_STRIDE + k] = (_Float16)v.x;
                ldsB[(c4 + 1) * LDS_STRIDE + k] = (_Float16)v.y;
                ldsB[(c4 + 2) * LDS_STRIDE + k] = (_Float16)v.z;
                ldsB[(c4 + 3) * LDS_STRIDE + k] = (_Float16)v.w;
            }
        } else {
            float vb[NBq];
#pragma unroll
            for (int u = 0; u < NBq; ++u) {
                int i = tid + u * 128;
                int r = i >> 6, c = i & 63;         // r = k, c = n
                int gk = min(kb + r, K - 1), gn = min(tileN + c, N - 1);
                vb[u] = B[(size_t)gk * N + gn];
            }
#pragma unroll
            for (int u = 0; u < NBq; ++u) {
                int i = tid + u * 128;
                int r = i >> 6, c = i & 63;
                bool ok = (kb + r < K) && (tileN + c < N);
                ldsB[c * LDS_STRIDE + r] = ok ? (_Float16)vb[u] : (_Float16)0.0f;
            }
        }

        // prefetch next A tile slice (global_prefetch_b8)
        if (kb + TILE_K < K) {
            int r = tid >> 1, c = (tid & 1) * 16;
            int gm = tileM + r, gk = kb + TILE_K + c;
            if (gm < M && gk < K) __builtin_prefetch(&A[(size_t)gm * K + gk], 0, 0);
        }
        __syncthreads();

        // ---- build fragments per CDNA5 16-bit WMMA VGPR layouts ----
        v16h afrag[2], bfrag[2];
        const int koffA = (lane & 16) ? 8 : 0;    // A: lanes>=16 hold K+8 block
        const int koffB = (lane & 16) ? 8 : 0;    // B: lanes>=16 hold K+8 block
        const int l15   = lane & 15;
#pragma unroll
        for (int s = 0; s < 2; ++s) {
            int ml = wm + s * 16 + l15;
            h8v alo = *(const h8v*)&ldsA[ml * LDS_STRIDE + koffA];
            h8v ahi = *(const h8v*)&ldsA[ml * LDS_STRIDE + koffA + 16];
#pragma unroll
            for (int e = 0; e < 8; ++e) { afrag[s][e] = alo[e]; afrag[s][e + 8] = ahi[e]; }
            int nl = wn + s * 16 + l15;
            h8v blo = *(const h8v*)&ldsB[nl * LDS_STRIDE + koffB];
            h8v bhi = *(const h8v*)&ldsB[nl * LDS_STRIDE + koffB + 16];
#pragma unroll
            for (int e = 0; e < 8; ++e) { bfrag[s][e] = blo[e]; bfrag[s][e + 8] = bhi[e]; }
        }
#pragma unroll
        for (int i = 0; i < 2; ++i)
#pragma unroll
            for (int j = 0; j < 2; ++j)
                acc[i][j] = __builtin_amdgcn_wmma_f32_16x16x32_f16(
                    false, afrag[i], false, bfrag[j], (short)0, acc[i][j], false, false);
        __syncthreads();
    }

    // ---- epilogue / store (C/D layout: N = lane&15, M = r + 8*(lane>=16)) ----
#pragma unroll
    for (int i = 0; i < 2; ++i) {
#pragma unroll
        for (int j = 0; j < 2; ++j) {
            int n = tileN + wn + j * 16 + (lane & 15);
            if (n >= N) continue;
            float bv = bias ? bias[n] : 0.0f;
            float gv = bng  ? bng[n]  : 1.0f;
            float b2 = bnb  ? bnb[n]  : 0.0f;
#pragma unroll
            for (int r = 0; r < 8; ++r) {
                int m = tileM + wm + i * 16 + r + ((lane & 16) ? 8 : 0);
                if (m >= M) continue;
                size_t idx = (size_t)m * N + n;
                float v = acc[i][j][r];
                if (Cprev) {
                    float s = 1.0f;
                    if (deg) s = 1.0f / fmaxf(deg[m], 1.0f);
                    v += Cprev[idx] * s;
                }
                v += bv;
                if (bng) v = v * gv + b2;
                if (elu) v = elu1(v);
                C[idx] = v;
            }
        }
    }
}

// ---------------------------------------------------------------------------
// Edge / graph kernels
// ---------------------------------------------------------------------------
__global__ void init_edges(const int* __restrict__ ei, const float* __restrict__ ea,
                           int* __restrict__ src, int* __restrict__ dst,
                           int* __restrict__ mask, float* __restrict__ ps, int E)
{
    int e = blockIdx.x * blockDim.x + threadIdx.x;
    if (e >= E) return;
    src[e] = ei[e];
    dst[e] = ei[E + e];
    mask[e] = 1;
    ps[2 * e]     = ea[2 * e];
    ps[2 * e + 1] = ea[2 * e + 1];
}

__global__ void degree_kernel(const int* __restrict__ dst, const int* __restrict__ mask,
                              float* __restrict__ deg, int E)
{
    int e = blockIdx.x * blockDim.x + threadIdx.x;
    if (e >= E) return;
    if (mask[e]) atomicAdd(&deg[dst[e]], 1.0f);
}

__device__ __forceinline__ void spline_basis(const float* pseudo, int e, int ks,
                                             int& iu0, int& iw0, int& iu1, int& iw1,
                                             float& fu, float& fw)
{
    float u = fminf(fmaxf(pseudo[2 * e],     0.0f), 1.0f) * (float)(ks - 1);
    float w = fminf(fmaxf(pseudo[2 * e + 1], 0.0f), 1.0f) * (float)(ks - 1);
    iu0 = (int)floorf(u);
    iw0 = (int)floorf(w);
    fu = u - (float)iu0;
    fw = w - (float)iw0;
    iu1 = min(iu0 + 1, ks - 1);
    iw1 = min(iw0 + 1, ks - 1);
}

// f32 bucket scatter (cin == 1 path, conv1)
__global__ void spline_scatter_f32(const float* __restrict__ X, const float* __restrict__ pseudo,
                                   const int* __restrict__ src, const int* __restrict__ dst,
                                   const int* __restrict__ mask, float* __restrict__ T,
                                   int E, int ks, int K)
{
    int e = blockIdx.x * blockDim.x + threadIdx.x;
    if (e >= E) return;
    if (!mask[e]) return;
    int iu0, iw0, iu1, iw1; float fu, fw;
    spline_basis(pseudo, e, ks, iu0, iw0, iu1, iw1, fu, fw);
    float xj = X[src[e]];
    size_t base = (size_t)dst[e] * K;
    atomicAdd(&T[base + (size_t)(iu0 * ks + iw0)], (1.0f - fu) * (1.0f - fw) * xj);
    atomicAdd(&T[base + (size_t)(iu0 * ks + iw1)], (1.0f - fu) * fw * xj);
    atomicAdd(&T[base + (size_t)(iu1 * ks + iw0)], fu * (1.0f - fw) * xj);
    atomicAdd(&T[base + (size_t)(iu1 * ks + iw1)], fu * fw * xj);
}

// packed f16 bucket scatter: one global_atomic_pk_add_f16 per channel pair
__global__ void spline_scatter_f16(const float* __restrict__ X, const float* __restrict__ pseudo,
                                   const int* __restrict__ src, const int* __restrict__ dst,
                                   const int* __restrict__ mask, _Float16* __restrict__ T,
                                   long long total, int cin2, int cin, int ks, int K)
{
    long long gid = (long long)blockIdx.x * blockDim.x + threadIdx.x;
    if (gid >= total) return;
    int e  = (int)(gid / cin2);
    int ch = (int)(gid - (long long)e * cin2) * 2;
    if (!mask[e]) return;
    int iu0, iw0, iu1, iw1; float fu, fw;
    spline_basis(pseudo, e, ks, iu0, iw0, iu1, iw1, fu, fw);
    float x0 = X[(size_t)src[e] * cin + ch];
    float x1 = X[(size_t)src[e] * cin + ch + 1];
    size_t base = (size_t)dst[e] * K * cin + ch;     // even -> 4B aligned
    float w00 = (1.0f - fu) * (1.0f - fw), w01 = (1.0f - fu) * fw;
    float w10 = fu * (1.0f - fw),          w11 = fu * fw;
    atomic_pk_add_f16(&T[base + (size_t)(iu0 * ks + iw0) * cin], w00 * x0, w00 * x1);
    atomic_pk_add_f16(&T[base + (size_t)(iu0 * ks + iw1) * cin], w01 * x0, w01 * x1);
    atomic_pk_add_f16(&T[base + (size_t)(iu1 * ks + iw0) * cin], w10 * x0, w10 * x1);
    atomic_pk_add_f16(&T[base + (size_t)(iu1 * ks + iw1) * cin], w11 * x0, w11 * x1);
}

__global__ void add_elu(const float* __restrict__ a, const float* __restrict__ b,
                        float* __restrict__ out, long long n)
{
    long long i = (long long)blockIdx.x * blockDim.x + threadIdx.x;
    if (i >= n) return;
    out[i] = elu1(a[i] + b[i]);
}

// ---------------------------------------------------------------------------
// Pooling kernels
// ---------------------------------------------------------------------------
__global__ void cluster_assign(const float* __restrict__ pos, const int* __restrict__ batch,
                               int* __restrict__ c, int n, float size, int g)
{
    int i = blockIdx.x * blockDim.x + threadIdx.x;
    if (i >= n) return;
    int cx = (int)floorf(pos[2 * i]     / size);
    int cy = (int)floorf(pos[2 * i + 1] / size);
    cx = min(max(cx, 0), g - 1);
    cy = min(max(cy, 0), g - 1);
    c[i] = batch[i] * (g * g) + cx * g + cy;
}

__global__ void pool_pos_cnt(const float* __restrict__ pos, const int* __restrict__ c,
                             float* __restrict__ cnt, float* __restrict__ psum, int n)
{
    int i = blockIdx.x * blockDim.x + threadIdx.x;
    if (i >= n) return;
    int cc = c[i];
    atomicAdd(&cnt[cc], 1.0f);
    atomicAdd(&psum[2 * cc],     pos[2 * i]);
    atomicAdd(&psum[2 * cc + 1], pos[2 * i + 1]);
}

__global__ void pool_max(const float* __restrict__ X, const int* __restrict__ c,
                         unsigned* __restrict__ xm, long long total, int C)
{
    long long gid = (long long)blockIdx.x * blockDim.x + threadIdx.x;
    if (gid >= total) return;
    int i  = (int)(gid / C);
    int ch = (int)(gid - (long long)i * C);
    atomicMax(&xm[(size_t)c[i] * C + ch], fenc(X[gid]));
}

__global__ void pool_finalize(unsigned* __restrict__ xm, const float* __restrict__ cnt,
                              const float* __restrict__ psum, float* __restrict__ posn,
                              int* __restrict__ batn, long long total, int C, int gg)
{
    long long gid = (long long)blockIdx.x * blockDim.x + threadIdx.x;
    if (gid >= total) return;
    int i = (int)(gid / C);
    float v = (cnt[i] > 0.0f) ? fdec(xm[gid]) : 0.0f;
    ((float*)xm)[gid] = v;     // in-place uint -> float
    if ((int)(gid - (long long)i * C) == 0) {
        float cc = fmaxf(cnt[i], 1.0f);
        posn[2 * i]     = psum[2 * i]     / cc;
        posn[2 * i + 1] = psum[2 * i + 1] / cc;
        batn[i] = i / gg;
    }
}

__global__ void edge_remap(int* __restrict__ src, int* __restrict__ dst,
                           int* __restrict__ mask, const int* __restrict__ c,
                           const float* __restrict__ posn, float* __restrict__ delta,
                           int E, unsigned* __restrict__ gmax)
{
    __shared__ float smax[256];
    int t = threadIdx.x;
    int e = blockIdx.x * 256 + t;
    float local = 0.0f;
    if (e < E) {
        int sn = c[src[e]], dn = c[dst[e]];
        int m = (mask[e] && (sn != dn)) ? 1 : 0;
        src[e] = sn; dst[e] = dn; mask[e] = m;
        float dx = 0.0f, dy = 0.0f;
        if (m) {
            dx = posn[2 * dn]     - posn[2 * sn];
            dy = posn[2 * dn + 1] - posn[2 * sn + 1];
        }
        delta[2 * e] = dx; delta[2 * e + 1] = dy;
        local = fmaxf(fabsf(dx), fabsf(dy));
    }
    smax[t] = local;
    __syncthreads();
    for (int o = 128; o > 0; o >>= 1) {
        if (t < o) smax[t] = fmaxf(smax[t], smax[t + o]);
        __syncthreads();
    }
    if (t == 0) atomicMax(gmax, fenc(smax[0]));
}

__global__ void edge_pseudo(float* __restrict__ delta, long long n,
                            const unsigned* __restrict__ gmax)
{
    long long i = (long long)blockIdx.x * blockDim.x + threadIdx.x;
    if (i >= n) return;
    float sc = 1.0f / (2.0f * fdec(*gmax) + 1e-12f);
    delta[i] = delta[i] * sc + 0.5f;
}

// ---------------------------------------------------------------------------
// fc2 + log_softmax (tiny: 8x1024 @ 1024x10)
// ---------------------------------------------------------------------------
__global__ void fc2_logsoftmax(const float* __restrict__ z, const float* __restrict__ W,
                               const float* __restrict__ b, float* __restrict__ out)
{
    int g = blockIdx.x;         // graph 0..7
    int t = threadIdx.x;        // 64 threads
    __shared__ float lg[10];
    if (t < 10) {
        float acc = b[t];
        const float* zr = z + (size_t)g * 1024;
        for (int k = 0; k < 1024; ++k) acc += zr[k] * W[k * 10 + t];
        lg[t] = acc;
    }
    __syncthreads();
    if (t == 0) {
        float mx = lg[0];
        for (int i = 1; i < 10; ++i) mx = fmaxf(mx, lg[i]);
        float s = 0.0f;
        for (int i = 0; i < 10; ++i) s += __expf(lg[i] - mx);
        float lse = mx + __logf(s);
        for (int i = 0; i < 10; ++i) out[g * 10 + i] = lg[i] - lse;
    }
}

// ---------------------------------------------------------------------------
// Host orchestration
// ---------------------------------------------------------------------------
namespace {
struct ConvP  { const float *w, *root, *b; };
struct BnP    { const float *g, *b; };
struct BlockP { ConvP l1; BnP bn1; ConvP l2; BnP bn2; ConvP sc; BnP bns; };
}

extern "C" void kernel_launch(void* const* d_in, const int* in_sizes, int n_in,
                              void* d_out, int out_size, void* d_ws, size_t ws_size,
                              hipStream_t stream)
{
    (void)in_sizes; (void)n_in; (void)out_size; (void)ws_size;
    constexpr int NB = 8, NPG = 16384, EPG = 131072;
    constexpr int N0 = NB * NPG;       // 131072 nodes
    constexpr int ET = NB * EPG;       // 1048576 edges (constant across levels)

    const float* x_in   = (const float*)d_in[0];
    const float* pos_in = (const float*)d_in[1];
    const float* ea_in  = (const float*)d_in[2];
    const int*   ei_in  = (const int*)d_in[3];
    const int*   bat_in = (const int*)d_in[4];

    int pi = 5;
    auto FPp = [&](int i) { return (const float*)d_in[i]; };
    auto getConv  = [&]() { ConvP c{FPp(pi), FPp(pi + 1), FPp(pi + 2)}; pi += 3; return c; };
    auto getBn    = [&]() { BnP b{FPp(pi), FPp(pi + 1)}; pi += 2; return b; };
    auto getBlock = [&]() { BlockP b; b.l1 = getConv(); b.bn1 = getBn();
                            b.l2 = getConv(); b.bn2 = getBn();
                            b.sc = getConv(); b.bns = getBn(); return b; };
    ConvP conv1 = getConv(); BnP bn1 = getBn();
    BlockP B1 = getBlock(), B2 = getBlock(), B3 = getBlock();
    const float* fc1w = FPp(pi++); const float* fc1b = FPp(pi++);
    const float* fc2w = FPp(pi++); const float* fc2b = FPp(pi++);

    // ---- workspace bump allocator ----
    size_t off = 0;
    auto alloc = [&](size_t bytes) {
        off = (off + 255) & ~(size_t)255;
        void* p = (char*)d_ws + off;
        off += bytes;
        return p;
    };
    const size_t FCAP = (size_t)N0 * 64;                 // 8,388,608 floats per feat buf
    float* F0 = (float*)alloc(FCAP * 4);
    float* F1 = (float*)alloc(FCAP * 4);
    float* F2 = (float*)alloc(FCAP * 4);
    float* F3 = (float*)alloc(FCAP * 4);
    void*  Tb = alloc((size_t)32768 * 25 * 128 * 2 + 256);  // max T (f16): 210 MB
    float* deg   = (float*)alloc((size_t)N0 * 4);
    int* srcb    = (int*)alloc((size_t)ET * 4);
    int* dstb    = (int*)alloc((size_t)ET * 4);
    int* maskb   = (int*)alloc((size_t)ET * 4);
    float* pseudo = (float*)alloc((size_t)ET * 2 * 4);
    float* posA  = (float*)alloc((size_t)32768 * 2 * 4);
    float* posB  = (float*)alloc((size_t)32768 * 2 * 4);
    int* batA    = (int*)alloc((size_t)32768 * 4);
    int* batB    = (int*)alloc((size_t)32768 * 4);
    int* clus    = (int*)alloc((size_t)N0 * 4);
    float* cnt   = (float*)alloc((size_t)32768 * 4);
    float* psum  = (float*)alloc((size_t)32768 * 2 * 4);
    unsigned* gmax = (unsigned*)alloc(256);
    float* z1    = (float*)alloc((size_t)8 * 1024 * 4);

    auto cdiv = [](long long a, long long b) { return (unsigned)((a + b - 1) / b); };

    auto gemmF32 = [&](const float* A, const float* Bm, float* C, int M, int N, int K,
                       const float* Cprev, const float* dg, const float* bias,
                       const float* g, const float* bb, int elu) {
        dim3 gr(cdiv(N, TILE_N), cdiv(M, TILE_M));
        wmma_gemm<float><<<gr, 128, 0, stream>>>(A, Bm, C, M, N, K, Cprev, dg, bias, g, bb, elu);
    };
    auto gemmF16 = [&](const _Float16* A, const float* Bm, float* C, int M, int N, int K,
                       const float* Cprev, const float* dg, const float* bias,
                       const float* g, const float* bb, int elu) {
        dim3 gr(cdiv(N, TILE_N), cdiv(M, TILE_M));
        wmma_gemm<_Float16><<<gr, 128, 0, stream>>>(A, Bm, C, M, N, K, Cprev, dg, bias, g, bb, elu);
    };

    // spline conv = scatter -> (T@W) -> (x@root + T@W/deg + bias; BN; ELU?)
    auto spline = [&](const float* X, int n, int cin, int cout, int ksz,
                      const ConvP& cv, const BnP& bn, int elu, float* OUT, float* TMP) {
        int Kk = ksz * ksz;
        if (cin == 1) {
            hipMemsetAsync(Tb, 0, (size_t)n * Kk * sizeof(float), stream);
            spline_scatter_f32<<<cdiv(ET, 256), 256, 0, stream>>>(
                X, pseudo, srcb, dstb, maskb, (float*)Tb, ET, ksz, Kk);
            gemmF32((const float*)Tb, cv.w, TMP, n, cout, Kk,
                    nullptr, nullptr, nullptr, nullptr, nullptr, 0);
        } else {
            hipMemsetAsync(Tb, 0, (size_t)n * Kk * cin * sizeof(_Float16), stream);
            long long tot = (long long)ET * (cin / 2);
            spline_scatter_f16<<<cdiv(tot, 256), 256, 0, stream>>>(
                X, pseudo, srcb, dstb, maskb, (_Float16*)Tb, tot, cin / 2, cin, ksz, Kk);
            gemmF16((const _Float16*)Tb, cv.w, TMP, n, cout, Kk * cin,
                    nullptr, nullptr, nullptr, nullptr, nullptr, 0);
        }
        gemmF32(X, cv.root, OUT, n, cout, cin, TMP, deg, cv.b, bn.g, bn.b, elu);
    };

    auto runBlock = [&](const BlockP& bp, float* X, int n, int cin, int cout,
                        float* Bh, float* Ch, float* Tmp) {
        spline(X,  n, cin,  cout, 5, bp.l1, bp.bn1, 1, Bh, Tmp);   // h1 (ELU'd)
        spline(Bh, n, cout, cout, 5, bp.l2, bp.bn2, 0, Ch, Tmp);   // h2
        spline(X,  n, cin,  cout, 1, bp.sc, bp.bns, 0, Bh, Tmp);   // shortcut
        long long tot = (long long)n * cout;
        add_elu<<<cdiv(tot, 256), 256, 0, stream>>>(Ch, Bh, X, tot); // out in X
    };

    auto pool = [&](const float* X, int n, int C, const float* posCur, const int* batCur,
                    float size, int g, float* XN, float* posNext, int* batNext, bool remap) {
        int ns = NB * g * g;
        cluster_assign<<<cdiv(n, 256), 256, 0, stream>>>(posCur, batCur, clus, n, size, g);
        hipMemsetAsync(cnt, 0, (size_t)ns * 4, stream);
        hipMemsetAsync(psum, 0, (size_t)ns * 2 * 4, stream);
        hipMemsetAsync(XN, 0, (size_t)ns * C * 4, stream);   // 0 == fenc minimum
        pool_pos_cnt<<<cdiv(n, 256), 256, 0, stream>>>(posCur, clus, cnt, psum, n);
        long long tn = (long long)n * C;
        pool_max<<<cdiv(tn, 256), 256, 0, stream>>>(X, clus, (unsigned*)XN, tn, C);
        long long ts = (long long)ns * C;
        pool_finalize<<<cdiv(ts, 256), 256, 0, stream>>>(
            (unsigned*)XN, cnt, psum, posNext, batNext, ts, C, g * g);
        if (remap) {
            hipMemsetAsync(gmax, 0, 4, stream);
            edge_remap<<<cdiv(ET, 256), 256, 0, stream>>>(
                srcb, dstb, maskb, clus, posNext, pseudo, ET, gmax);
            edge_pseudo<<<cdiv((long long)2 * ET, 256), 256, 0, stream>>>(
                pseudo, (long long)2 * ET, gmax);
            hipMemsetAsync(deg, 0, (size_t)ns * 4, stream);
            degree_kernel<<<cdiv(ET, 256), 256, 0, stream>>>(dstb, maskb, deg, ET);
        }
    };

    // ================= pipeline =================
    init_edges<<<cdiv(ET, 256), 256, 0, stream>>>(ei_in, ea_in, srcb, dstb, maskb, pseudo, ET);
    hipMemsetAsync(deg, 0, (size_t)N0 * 4, stream);
    degree_kernel<<<cdiv(ET, 256), 256, 0, stream>>>(dstb, maskb, deg, ET);

    // conv1: [131072,1] -> [131072,64], ELU
    spline(x_in, N0, 1, 64, 5, conv1, bn1, 1, F0, F3);

    // pool(4): g=64 -> 32768 clusters
    pool(F0, N0, 64, pos_in, bat_in, 4.0f, 64, F1, posA, batA, true);
    // block1: 64 -> 128 (out in F1)
    runBlock(B1, F1, 32768, 64, 128, F2, F0, F3);

    // pool(6): g=43 -> 14792 clusters
    pool(F1, 32768, 128, posA, batA, 6.0f, 43, F0, posB, batB, true);
    // block2: 128 -> 256 (out in F0)
    runBlock(B2, F0, 14792, 128, 256, F1, F2, F3);

    // pool(20): g=13 -> 1352 clusters
    pool(F0, 14792, 256, posB, batB, 20.0f, 13, F1, posA, batA, true);
    // block3: 256 -> 512 (out in F1)
    runBlock(B3, F1, 1352, 256, 512, F2, F0, F3);

    // final pool(32): g=8 -> 512 clusters == z[8, 32768]
    pool(F1, 1352, 512, posA, batA, 32.0f, 8, F2, posB, batB, false);

    // fc1: [8,32768] @ [32768,1024] + b, ELU
    gemmF32(F2, fc1w, z1, 8, 1024, 32768, nullptr, nullptr, fc1b, nullptr, nullptr, 1);
    // fc2 + log_softmax -> d_out [8,10]
    fc2_logsoftmax<<<8, 64, 0, stream>>>(z1, fc2w, fc2b, (float*)d_out);
}
